// RWKV_TimeMix_54185307407105
// MI455X (gfx1250) — matrix operand order repaired and checked by reading, not compiled, not run on hardware
//
#include <hip/hip_runtime.h>

typedef float v2f __attribute__((ext_vector_type(2)));
typedef float v4f __attribute__((ext_vector_type(4)));
typedef float v8f __attribute__((ext_vector_type(8)));
typedef __bf16 v8bf __attribute__((ext_vector_type(8)));
typedef __bf16 v16bf __attribute__((ext_vector_type(16)));

#define T_DIM 1024
#define C_DIM 1024
#define H_DIM 16
#define HS_DIM 64
#define B_DIM 8
#define NEL (B_DIM * T_DIM * C_DIM)

// ---------------------------------------------------------------------------
// Async global->LDS DMA (CDNA5, ASYNCcnt). Per-lane 16B transfer.
// ---------------------------------------------------------------------------
__device__ __forceinline__ void async_b128(unsigned lds_off, const void* gptr) {
  asm volatile("global_load_async_to_lds_b128 %0, %1, off" ::"v"(lds_off),
               "v"((unsigned long long)(uintptr_t)gptr)
               : "memory");
}
__device__ __forceinline__ void wait_async0() {
  asm volatile("s_wait_asynccnt 0" ::: "memory");
}

// ---------------------------------------------------------------------------
// fp32 -> bf16(hi) + bf16(lo) split (round-to-nearest-even)
// ---------------------------------------------------------------------------
__device__ __forceinline__ unsigned short bf16_rne(float f) {
  unsigned u = __float_as_uint(f);
  return (unsigned short)((u + 0x7FFFu + ((u >> 16) & 1u)) >> 16);
}
__device__ __forceinline__ void split_bf16(float f, unsigned short& h,
                                           unsigned short& l) {
  h = bf16_rne(f);
  float fh = __uint_as_float((unsigned)h << 16);
  l = bf16_rne(f - fh);
}

__global__ __launch_bounds__(256) void split_kernel(
    const float* __restrict__ src, unsigned short* __restrict__ hi,
    unsigned short* __restrict__ lo) {
  size_t idx = (size_t)blockIdx.x * 256 + threadIdx.x;
  unsigned short h, l;
  split_bf16(src[idx], h, l);
  hi[idx] = h;
  lo[idx] = l;
}

__global__ __launch_bounds__(256) void build_xs_split_kernel(
    const float* __restrict__ x, unsigned short* __restrict__ hi,
    unsigned short* __restrict__ lo) {
  size_t idx = (size_t)blockIdx.x * 256 + threadIdx.x;
  int c = (int)(idx & (C_DIM - 1));
  int t = (int)((idx >> 10) & (T_DIM - 1));
  float v = (c < C_DIM / 2) ? ((t > 0) ? x[idx - C_DIM] : 0.0f) : x[idx];
  unsigned short h, l;
  split_bf16(v, h, l);
  hi[idx] = h;
  lo[idx] = l;
}

// ---------------------------------------------------------------------------
// bf16-split WMMA GEMM: out[m,n] = epi( sum_k A[m,k]*W[n,k] + bias[n] )
// 128 threads = 4 waves. Block tile 128(M) x 64(N); wave tile 32 x 64.
// Double-buffered LDS fed by global_load_async_to_lds_b128.
// EPI: 0 none, 1 exp, 2 sigmoid, 3 *gamma[row%T]
// ---------------------------------------------------------------------------
#define BKB 32
#define SAH 40 /* LDS row stride (bf16 elems) */
#define NCH (C_DIM / BKB)
#define A_PLANE (128 * SAH)           /* elems */
#define B_PLANE (64 * SAH)            /* elems */
#define STAGE_ELEMS (2 * A_PLANE + 2 * B_PLANE)

__device__ __forceinline__ v16bf frag16(const unsigned short* p0,
                                        const unsigned short* p1) {
  v8bf a = *(const v8bf*)p0;
  v8bf b = *(const v8bf*)p1;
  return __builtin_shufflevector(a, b, 0, 1, 2, 3, 4, 5, 6, 7, 8, 9, 10, 11,
                                 12, 13, 14, 15);
}

template <int EPI>
__global__ __launch_bounds__(128) void gemm_bf16_kernel(
    const unsigned short* __restrict__ Ah, const unsigned short* __restrict__ Al,
    const unsigned short* __restrict__ Wh, const unsigned short* __restrict__ Wl,
    const float* __restrict__ bias, float* __restrict__ out,
    const float* __restrict__ gamma) {
  __shared__ unsigned short smem[2][STAGE_ELEMS];

  const int tid = threadIdx.x;
  const int wave = tid >> 5, lane = tid & 31, m16 = lane & 15, hi = lane >> 4;
  const int rowBase = blockIdx.y * 128, colBase = blockIdx.x * 64;
  const unsigned lds_base = (unsigned)(uintptr_t)(void*)&smem[0][0];

  // Issue async DMA of one K-chunk into stage st. 12 x b128 per thread.
  auto stage_async = [&](int k0, int st) {
    const unsigned sbase = lds_base + (unsigned)(st * STAGE_ELEMS * 2);
#pragma unroll
    for (int i = 0; i < 4; ++i) {  // A planes: 512 b128 each
      int idx = tid + i * 128;
      int row = idx >> 2, kq = (idx & 3) << 3;
      unsigned loff = (unsigned)((row * SAH + kq) * 2);
      size_t g = (size_t)(rowBase + row) * C_DIM + k0 + kq;
      async_b128(sbase + loff, Ah + g);
      async_b128(sbase + A_PLANE * 2 + loff, Al + g);
    }
#pragma unroll
    for (int i = 0; i < 2; ++i) {  // W planes: 256 b128 each
      int idx = tid + i * 128;
      int row = idx >> 2, kq = (idx & 3) << 3;
      unsigned loff = (unsigned)((row * SAH + kq) * 2);
      size_t g = (size_t)(colBase + row) * C_DIM + k0 + kq;
      async_b128(sbase + 2 * A_PLANE * 2 + loff, Wh + g);
      async_b128(sbase + (2 * A_PLANE + B_PLANE) * 2 + loff, Wl + g);
    }
  };

  v8f acc[2][4];
#pragma unroll
  for (int mt = 0; mt < 2; ++mt)
#pragma unroll
    for (int j = 0; j < 4; ++j)
#pragma unroll
      for (int r = 0; r < 8; ++r) acc[mt][j][r] = 0.0f;

  stage_async(0, 0);
  wait_async0();
  __syncthreads();

  for (int ch = 0; ch < NCH; ++ch) {
    const int cur = ch & 1;
    const bool nxt = (ch + 1) < NCH;
    if (nxt) stage_async((ch + 1) * BKB, 1 - cur);

    const unsigned short* s0 = smem[cur];
    const unsigned short* sal = s0 + A_PLANE;
    const unsigned short* sbh = s0 + 2 * A_PLANE;
    const unsigned short* sbl = sbh + B_PLANE;

    // A frags: K runs [8*hi, +7] and [16+8*hi, +7] per ISA 16-bit A layout
    v16bf ahi[2], alo[2];
#pragma unroll
    for (int mt = 0; mt < 2; ++mt) {
      const int aoff = (wave * 32 + mt * 16 + m16) * SAH + 8 * hi;
      ahi[mt] = frag16(s0 + aoff, s0 + aoff + 16);
      alo[mt] = frag16(sal + aoff, sal + aoff + 16);
    }
#pragma unroll
    for (int j = 0; j < 4; ++j) {
      // B frag: K runs [8*hi, +7] and [16+8*hi, +7] (matches A layout)
      const int boff = (j * 16 + m16) * SAH + 8 * hi;
      v16bf bhi = frag16(sbh + boff, sbh + boff + 16);
      v16bf blo = frag16(sbl + boff, sbl + boff + 16);
#pragma unroll
      for (int mt = 0; mt < 2; ++mt) {
        acc[mt][j] = __builtin_amdgcn_wmma_f32_16x16x32_bf16(
            false, ahi[mt], false, bhi, (short)0, acc[mt][j], false, false);
        acc[mt][j] = __builtin_amdgcn_wmma_f32_16x16x32_bf16(
            false, ahi[mt], false, blo, (short)0, acc[mt][j], false, false);
        acc[mt][j] = __builtin_amdgcn_wmma_f32_16x16x32_bf16(
            false, alo[mt], false, bhi, (short)0, acc[mt][j], false, false);
      }
    }
    wait_async0();
    __syncthreads();
  }

  // Epilogue: element (row = rowBase + wave*32 + mt*16 + r + 8*hi,
  //                    col = colBase + j*16 + m16)
#pragma unroll
  for (int j = 0; j < 4; ++j) {
    int n = colBase + j * 16 + m16;
    float bv = bias[n];
#pragma unroll
    for (int mt = 0; mt < 2; ++mt) {
#pragma unroll
      for (int r = 0; r < 8; ++r) {
        int row = rowBase + wave * 32 + mt * 16 + r + 8 * hi;
        float v = acc[mt][j][r] + bv;
        if (EPI == 1) v = expf(v);
        if (EPI == 2) v = 1.0f / (1.0f + expf(-v));
        if (EPI == 3) v = v * gamma[row & (T_DIM - 1)];
        out[(size_t)row * C_DIM + n] = v;
      }
    }
  }
}

// ---------------------------------------------------------------------------
// kv_alpha: v <- alpha[h, t] * k * v   (in place over v)
// ---------------------------------------------------------------------------
__global__ __launch_bounds__(256) void kv_alpha_kernel(
    const float* __restrict__ k, float* __restrict__ v,
    const float* __restrict__ alpha) {
  size_t idx = (size_t)blockIdx.x * 256 + threadIdx.x;
  int c = (int)(idx & (C_DIM - 1));
  int t = (int)((idx >> 10) & (T_DIM - 1));
  int h = c >> 6;
  v[idx] = alpha[h * T_DIM + t] * k[idx] * v[idx];
}

// ---------------------------------------------------------------------------
// cumsum over t, in place on k. One thread per (b,c); coalesced in c.
// ---------------------------------------------------------------------------
__global__ __launch_bounds__(256) void cumsum_kernel(float* __restrict__ k) {
  int gid = blockIdx.x * 256 + threadIdx.x;
  int b = gid >> 10;
  int c = gid & (C_DIM - 1);
  float s = 0.0f;
  size_t base = (size_t)b * T_DIM * C_DIM + c;
  for (int t = 0; t < T_DIM; ++t) {
    size_t idx = base + (size_t)t * C_DIM;
    s += k[idx];
    k[idx] = s;
  }
}

// ---------------------------------------------------------------------------
// wkv: per (b, h, t-block of 128) causal matmul, fp32 16x16x4 WMMA.
// 128 threads = 4 waves; wave owns 32 t-rows (2 tiles) x 64 cols so each
// LDS B-fragment feeds 2 WMMAs. Decay A built branchlessly from time_w.
//   acc[t, c] = sum_{u<=t} time_w[h, 1023 - t + u] * a[b, u, h, c]
//   y = rsig * beta[h,t] * acc / sumk   (in place over rsig)
// ---------------------------------------------------------------------------
#define WSBS 34

__global__ __launch_bounds__(128) void wkv_kernel(
    const float* __restrict__ a,     // alpha*k*v  [B,T,C]
    const float* __restrict__ tw,    // time_w     [H,TT]
    const float* __restrict__ beta,  // time_beta  [H,TT]
    const float* __restrict__ sumk,  // cumsum(k)  [B,T,C]
    float* __restrict__ y) {         // in: rsig, out: y  [B,T,C]
  __shared__ float sb[64 * WSBS];

  const int tid = threadIdx.x;
  const int wave = tid >> 5, lane = tid & 31, m16 = lane & 15, hi = lane >> 4;
  const int tblk = blockIdx.x, h = blockIdx.y, b = blockIdx.z;

  const int t0w = tblk * 128 + wave * 32;
  const int tmax = t0w + 31;
  const float* twh = tw + h * T_DIM;

  v8f acc[2][4];
#pragma unroll
  for (int mt = 0; mt < 2; ++mt)
#pragma unroll
    for (int j = 0; j < 4; ++j)
#pragma unroll
      for (int r = 0; r < 8; ++r) acc[mt][j][r] = 0.0f;

  const int nchunk = tblk * 4 + 4;
  for (int ch = 0; ch < nchunk; ++ch) {
    const int u0 = ch * 32;
    // Stage a[b, u0..u0+31, h*64..+63] transposed -> sb[c][u]
#pragma unroll
    for (int i = 0; i < 4; ++i) {
      int idx = tid + i * 128;
      int uu = idx >> 4;
      int cq = (idx & 15) << 2;
      v4f vec = *(const v4f*)(a + ((size_t)(b * T_DIM + u0 + uu) * C_DIM) +
                              h * HS_DIM + cq);
      sb[(cq + 0) * WSBS + uu] = vec.x;
      sb[(cq + 1) * WSBS + uu] = vec.y;
      sb[(cq + 2) * WSBS + uu] = vec.z;
      sb[(cq + 3) * WSBS + uu] = vec.w;
    }
    __syncthreads();

    if (u0 <= tmax) {
#pragma unroll
      for (int s = 0; s < 8; ++s) {
        const int kk = 4 * s + 2 * hi;
        const int ux = u0 + kk;
        v2f bfr[4];
#pragma unroll
        for (int j = 0; j < 4; ++j)
          bfr[j] = *(const v2f*)(sb + (j * 16 + m16) * WSBS + kk);
#pragma unroll
        for (int mt = 0; mt < 2; ++mt) {
          const int trow = t0w + mt * 16 + m16;
          int d0 = 1023 - trow + ux;
          int d1 = d0 + 1;
          d0 = d0 > 1023 ? 1023 : d0;
          d1 = d1 > 1023 ? 1023 : d1;
          float w0 = twh[d0];
          float w1 = twh[d1];
          v2f af;
          af.x = (ux <= trow) ? w0 : 0.0f;
          af.y = (ux + 1 <= trow) ? w1 : 0.0f;
#pragma unroll
          for (int j = 0; j < 4; ++j)
            acc[mt][j] = __builtin_amdgcn_wmma_f32_16x16x4_f32(
                false, af, false, bfr[j], (short)0, acc[mt][j], false, false);
        }
      }
    }
    __syncthreads();
  }

#pragma unroll
  for (int mt = 0; mt < 2; ++mt) {
#pragma unroll
    for (int j = 0; j < 4; ++j) {
#pragma unroll
      for (int r = 0; r < 8; ++r) {
        int t = t0w + mt * 16 + r + 8 * hi;
        int c = j * 16 + m16;
        size_t idx = (size_t)(b * T_DIM + t) * C_DIM + h * HS_DIM + c;
        float rs = y[idx];  // sigmoid(r)
        y[idx] = rs * beta[h * T_DIM + t] * acc[mt][j][r] / sumk[idx];
      }
    }
  }
}

// ---------------------------------------------------------------------------
extern "C" void kernel_launch(void* const* d_in, const int* in_sizes, int n_in,
                              void* d_out, int out_size, void* d_ws,
                              size_t ws_size, hipStream_t stream) {
  const float* x     = (const float*)d_in[0];
  const float* tw    = (const float*)d_in[1];
  const float* alpha = (const float*)d_in[2];
  const float* beta  = (const float*)d_in[3];
  const float* gamma = (const float*)d_in[4];
  const float* Wk    = (const float*)d_in[5];
  const float* bk    = (const float*)d_in[6];
  const float* Wv    = (const float*)d_in[7];
  const float* bv    = (const float*)d_in[8];
  const float* Wr    = (const float*)d_in[9];
  const float* br    = (const float*)d_in[10];
  const float* Wo    = (const float*)d_in[11];
  const float* bo    = (const float*)d_in[12];
  float* out = (float*)d_out;

  char* ws = (char*)d_ws;
  const size_t MB = 1 << 20;
  unsigned short* xs_hi = (unsigned short*)(ws + 0 * MB);   // 16MB (reused as y_hi)
  unsigned short* xs_lo = (unsigned short*)(ws + 16 * MB);  // 16MB (reused as y_lo)
  unsigned short* wk_hi = (unsigned short*)(ws + 32 * MB);
  unsigned short* wk_lo = (unsigned short*)(ws + 34 * MB);
  unsigned short* wv_hi = (unsigned short*)(ws + 36 * MB);
  unsigned short* wv_lo = (unsigned short*)(ws + 38 * MB);
  unsigned short* wr_hi = (unsigned short*)(ws + 40 * MB);
  unsigned short* wr_lo = (unsigned short*)(ws + 42 * MB);
  unsigned short* wo_hi = (unsigned short*)(ws + 44 * MB);
  unsigned short* wo_lo = (unsigned short*)(ws + 46 * MB);
  float* kb = (float*)(ws + 48 * MB);
  float* vb = (float*)(ws + 80 * MB);
  float* rb = (float*)(ws + 112 * MB);  // total 144MB

  const dim3 blk256(256);
  const dim3 blk128(128);
  const dim3 gridEW(NEL / 256);                              // 32768
  const dim3 gridW((C_DIM * C_DIM) / 256);                   // 4096
  const dim3 gridGemm(C_DIM / 64, (B_DIM * T_DIM) / 128);    // (16, 64)

  // 1. xs -> bf16 hi/lo planes
  build_xs_split_kernel<<<gridEW, blk256, 0, stream>>>(x, xs_hi, xs_lo);
  // 2. weights -> bf16 hi/lo planes
  split_kernel<<<gridW, blk256, 0, stream>>>(Wk, wk_hi, wk_lo);
  split_kernel<<<gridW, blk256, 0, stream>>>(Wv, wv_hi, wv_lo);
  split_kernel<<<gridW, blk256, 0, stream>>>(Wr, wr_hi, wr_lo);
  split_kernel<<<gridW, blk256, 0, stream>>>(Wo, wo_hi, wo_lo);
  // 3. k = exp(.), v = ., rsig = sigmoid(.)
  gemm_bf16_kernel<1><<<gridGemm, blk128, 0, stream>>>(xs_hi, xs_lo, wk_hi,
                                                       wk_lo, bk, kb, nullptr);
  gemm_bf16_kernel<0><<<gridGemm, blk128, 0, stream>>>(xs_hi, xs_lo, wv_hi,
                                                       wv_lo, bv, vb, nullptr);
  gemm_bf16_kernel<2><<<gridGemm, blk128, 0, stream>>>(xs_hi, xs_lo, wr_hi,
                                                       wr_lo, br, rb, nullptr);
  // 4. vb <- alpha * k * v ; kb <- cumsum(k)
  kv_alpha_kernel<<<gridEW, blk256, 0, stream>>>(kb, vb, alpha);
  cumsum_kernel<<<dim3((B_DIM * C_DIM) / 256), blk256, 0, stream>>>(kb);
  // 5. rb <- y (in place over rsig)
  wkv_kernel<<<dim3(T_DIM / 128, H_DIM, B_DIM), blk128, 0, stream>>>(
      vb, tw, beta, kb, rb);
  // 6. y -> bf16 planes (reuse xs planes), final GEMM with gamma epilogue
  split_kernel<<<gridEW, blk256, 0, stream>>>(rb, xs_hi, xs_lo);
  gemm_bf16_kernel<3><<<gridGemm, blk128, 0, stream>>>(xs_hi, xs_lo, wo_hi,
                                                       wo_lo, bo, out, gamma);
}
